// LinearModel_33268816675036
// MI455X (gfx1250) — matrix operand order repaired and checked by reading, not compiled, not run on hardware
//
#include <hip/hip_runtime.h>

// MI455X / gfx1250: wave32, WMMA bf16 16x16x32 (f32 accumulate).
// W1 pre-converted once to bf16 in WMMA-B order (52 MB in d_ws; whole W1
// working set lives in the 192 MB L2). Hot kernel: 64-row M-tile (one b per
// WG), double-buffered global_load_async_to_lds_b128 staging (ASYNCcnt)
// overlapped with 16 WMMAs per wave per K-tile.
typedef __attribute__((ext_vector_type(16))) __bf16 v16bf;
typedef __attribute__((ext_vector_type(8)))  float  v8f;
typedef __attribute__((ext_vector_type(4)))  int    i32x4;
typedef __attribute__((ext_vector_type(4)))  float  f32x4;
typedef __attribute__((ext_vector_type(2)))  unsigned int u32x2;

#define Bdim 32
#define Cdim 64
#define Edim 512
#define Ldim 100

#define ASTR 520  // A slab stride (elements): 1040B rows -> distinct bank groups
#define BSTR 40   // B tile stride (elements): 80B rows, conflict-free b128 reads
#define BBUF (512 * BSTR)          // one B buffer, elements
#define BTILE_BYTES 32768          // 512 n * 32 k * 2B, contiguous per K-tile

__device__ __forceinline__ unsigned short f2bf(float f) {
    unsigned int u = __builtin_bit_cast(unsigned int, f);
    unsigned int r = (u + 0x7FFFu + ((u >> 16) & 1u)) >> 16;  // round-to-nearest-even
    return (unsigned short)r;
}

// CDNA5 async global->LDS copies (ASYNCcnt). GV form: 64-bit vaddr.
__device__ __forceinline__ void async_b128(unsigned lds_byte_off, const void* gptr) {
    asm volatile("global_load_async_to_lds_b128 %0, %1, off"
                 :: "v"(lds_byte_off), "v"((unsigned long long)gptr) : "memory");
}
// GVS form: SGPR base + 32-bit per-lane offset (no per-issue VALU adds).
__device__ __forceinline__ void async_b128_gvs(unsigned lds_byte_off, unsigned voff,
                                               const void* sbase) {
    asm volatile("global_load_async_to_lds_b128 %0, %1, %2"
                 :: "v"(lds_byte_off), "v"(voff), "s"((unsigned long long)sbase)
                 : "memory");
}
__device__ __forceinline__ void wait_async0() {
    asm volatile("s_wait_asynccnt 0x0" ::: "memory");
}

union Frag { i32x4 q[2]; v16bf v; };

__global__ __launch_bounds__(256) void zero_y_kernel(float* __restrict__ out) {
    int i = blockIdx.x * blockDim.x + threadIdx.x;
    if (i < Bdim * Ldim) out[i] = 0.0f;
}

// fp32 -> bf16 convert + transpose of W1 into WMMA-B order:
// out[l][kt][n][k32] = W1[l][kt*32 + k32][n]  (each 32KB B-tile contiguous).
__global__ __launch_bounds__(256) void conv_w1_kernel(
    const float* __restrict__ W1, unsigned short* __restrict__ W1bf) {
    __shared__ __align__(16) unsigned short Tl[128 * 40];
    const int t  = threadIdx.x;
    const int l  = blockIdx.z;
    const int kt = blockIdx.y;          // 0..15
    const int nb = blockIdx.x * 128;    // 0,128,256,384
    const float* src = W1 + ((long)l * Edim + kt * 32) * Edim;
    #pragma unroll
    for (int i = 0; i < 4; ++i) {
        int q  = i * 256 + t;           // 0..1023
        int k  = q >> 5;                // 0..31
        int n4 = (q & 31) << 2;         // 0..124
        f32x4 v = *reinterpret_cast<const f32x4*>(&src[(long)k * Edim + nb + n4]);
        Tl[(n4 + 0) * 40 + k] = f2bf(v.x);
        Tl[(n4 + 1) * 40 + k] = f2bf(v.y);
        Tl[(n4 + 2) * 40 + k] = f2bf(v.z);
        Tl[(n4 + 3) * 40 + k] = f2bf(v.w);
    }
    __syncthreads();
    unsigned short* dst = W1bf + (((long)l * 16 + kt) * Edim + nb) * 32;
    #pragma unroll
    for (int i = 0; i < 2; ++i) {
        int c = i * 256 + t;            // 0..511
        int n = c >> 2, ch = (c & 3) << 3;
        i32x4 v = *reinterpret_cast<const i32x4*>(&Tl[n * 40 + ch]);
        *reinterpret_cast<i32x4*>(&dst[n * 32 + ch]) = v;
    }
}

// fp32 -> bf16 elementwise convert of c_emb (row-major kept).
__global__ __launch_bounds__(256) void conv_cemb_kernel(
    const float* __restrict__ src, unsigned short* __restrict__ dst) {
    long i = (long)blockIdx.x * 256 + threadIdx.x;   // quad index
    f32x4 v = *reinterpret_cast<const f32x4*>(src + i * 4);
    u32x2 o;
    o.x = (unsigned)f2bf(v.x) | ((unsigned)f2bf(v.y) << 16);
    o.y = (unsigned)f2bf(v.z) | ((unsigned)f2bf(v.w) << 16);
    *reinterpret_cast<u32x2*>(dst + i * 4) = o;
}

// ---------------- main pipelined kernel (PRECONV path) -----------------
// One WG = one b (64 rows) x one label l. 8 waves: wm in {0,1} covers two
// 16-row sub-tiles each; wn in {0..3} interleaves the 32 N-tiles.
__global__ __launch_bounds__(256) void mlp_pipelined_kernel(
    const unsigned short* __restrict__ cembbf,  // [B*C,E] bf16
    const float* __restrict__ c_pred,           // [B,C]
    const unsigned short* __restrict__ W1bf,    // [L][16][512][32] bf16
    const float* __restrict__ b1,               // [L,E]
    const float* __restrict__ W2,               // [L,E]
    const float* __restrict__ b2,               // [L]
    float* __restrict__ out)                    // [B*L] y ++ [B*C*L] logits
{
    __shared__ __align__(16) unsigned short Al[64 * ASTR];   // 66.5 KB, full A slab
    __shared__ __align__(16) unsigned short Bl[2 * BBUF];    // 80 KB, double buffer
    __shared__ float wl[64];
    __shared__ float ysum[2];

    const int t    = threadIdx.x;
    const int lane = t & 31;
    const int wave = t >> 5;
    const int wm   = wave >> 2;   // 0..1 : 32-row M group (2 sub-tiles of 16)
    const int wn   = wave & 3;    // 0..3 : N-tile interleave
    const int half = lane >> 4;   // 0..1
    const int l15  = lane & 15;

    const int l       = blockIdx.y;
    const int rowBase = blockIdx.x * 64;   // = bIdx * Cdim
    const int bIdx    = blockIdx.x;

    if (t < 64) wl[t] = 0.0f;

    const unsigned AlOff = (unsigned)(unsigned long long)(void*)Al;
    const unsigned BlOff = (unsigned)(unsigned long long)(void*)Bl;

    v8f acc0[8], acc1[8];
    #pragma unroll
    for (int j = 0; j < 8; ++j) { acc0[j] = (v8f)0.0f; acc1[j] = (v8f)0.0f; }

    // ---- prologue: async-load the whole 64x512 A slab (64 KB) once ----
    const char* aSrc = (const char*)(cembbf + (long)rowBase * Edim);
    #pragma unroll
    for (int i = 0; i < 16; ++i) {
        int c = i * 256 + t;            // 16B-chunk 0..4095
        int m = c >> 6;                 // row 0..63
        int chb = (c & 63) << 4;        // byte in row 0..1008
        async_b128(AlOff + (unsigned)(m * ASTR * 2) + (unsigned)chb,
                   aSrc + (long)m * (Edim * 2) + chb);
    }

    // ---- B staging: loop-invariant LDS dests + per-lane voffset ----
    const char* bBase = (const char*)(W1bf + (long)l * Edim * Edim);
    unsigned ldsB[8];
    #pragma unroll
    for (int i = 0; i < 8; ++i)
        ldsB[i] = BlOff + (unsigned)(((i * 64 + (t >> 2)) * BSTR + ((t & 3) << 3)) * 2);
    const unsigned voffB = (unsigned)(t * 16);

    // B tile 0 into buffer 0
    #pragma unroll
    for (int i = 0; i < 8; ++i)
        async_b128_gvs(ldsB[i], voffB, bBase + i * 4096);
    wait_async0();
    __syncthreads();

    const int m0 = wm * 32 + l15;       // sub-tile 0 row
    const int m1 = m0 + 16;             // sub-tile 1 row
    for (int kt = 0; kt < 16; ++kt) {
        // prefetch next B tile into the other buffer (overlaps with WMMA below)
        if (kt < 15) {
            const char* nb = bBase + (long)(kt + 1) * BTILE_BYTES;
            const unsigned bo = ((kt + 1) & 1) ? (unsigned)(BBUF * 2) : 0u;
            #pragma unroll
            for (int i = 0; i < 8; ++i)
                async_b128_gvs(ldsB[i] + bo, voffB, nb + i * 4096);
        }

        // compute on current buffer: 2 A frags x 8 B frags = 16 WMMAs
        const int k0 = kt * 32;
        Frag a0, a1;
        a0.q[0] = *reinterpret_cast<const i32x4*>(&Al[m0 * ASTR + k0 + half * 8]);
        a0.q[1] = *reinterpret_cast<const i32x4*>(&Al[m0 * ASTR + k0 + 16 + half * 8]);
        a1.q[0] = *reinterpret_cast<const i32x4*>(&Al[m1 * ASTR + k0 + half * 8]);
        a1.q[1] = *reinterpret_cast<const i32x4*>(&Al[m1 * ASTR + k0 + 16 + half * 8]);
        const unsigned short* Bcur = Bl + (kt & 1) * BBUF;
        #pragma unroll
        for (int j = 0; j < 8; ++j) {
            const int nt = (j * 4 + wn) * 16;
            Frag bfr;
            const i32x4* p = reinterpret_cast<const i32x4*>(&Bcur[(nt + l15) * BSTR + half * 8]);
            bfr.q[0] = p[0];
            bfr.q[1] = p[2];
            acc0[j] = __builtin_amdgcn_wmma_f32_16x16x32_bf16(
                false, a0.v, false, bfr.v, (short)0, acc0[j], false, false);
            acc1[j] = __builtin_amdgcn_wmma_f32_16x16x32_bf16(
                false, a1.v, false, bfr.v, (short)0, acc1[j], false, false);
        }

        wait_async0();      // next tile landed
        __syncthreads();    // everyone done reading current buffer
    }

    // ---- epilogue: h = relu(acc + b1[n]); w = sum_n h * W2[n] ----
    float ws0[8], ws1[8];
    #pragma unroll
    for (int r = 0; r < 8; ++r) { ws0[r] = 0.0f; ws1[r] = 0.0f; }
    #pragma unroll
    for (int j = 0; j < 8; ++j) {
        const int n  = (j * 4 + wn) * 16 + l15;
        const float bb = b1[l * Edim + n];
        const float ww = W2[l * Edim + n];
        #pragma unroll
        for (int r = 0; r < 8; ++r) {
            float v0 = acc0[j][r] + bb;
            float v1 = acc1[j][r] + bb;
            v0 = v0 > 0.0f ? v0 : 0.0f;
            v1 = v1 > 0.0f ? v1 : 0.0f;
            ws0[r] += v0 * ww;
            ws1[r] += v1 * ww;
        }
    }
    #pragma unroll
    for (int r = 0; r < 8; ++r) {
        float s0 = ws0[r], s1 = ws1[r];
        s0 += __shfl_xor(s0, 1, 32);  s1 += __shfl_xor(s1, 1, 32);
        s0 += __shfl_xor(s0, 2, 32);  s1 += __shfl_xor(s1, 2, 32);
        s0 += __shfl_xor(s0, 4, 32);  s1 += __shfl_xor(s1, 4, 32);
        s0 += __shfl_xor(s0, 8, 32);  s1 += __shfl_xor(s1, 8, 32);
        if (l15 == 0) {
            atomicAdd(&wl[wm * 32 + half * 8 + r], s0);       // ds_add_f32
            atomicAdd(&wl[wm * 32 + 16 + half * 8 + r], s1);
        }
    }
    __syncthreads();

    // 64 rows = exactly one b: logits written directly, y reduced in-WG
    float lg = 0.0f;
    if (t < 64) {
        const int row = rowBase + t;                   // row = b*C + c
        const float w = wl[t] + b2[l];
        lg = w * c_pred[row];
        out[Bdim * Ldim + (long)row * Ldim + l] = lg;  // logits[b,c,l]
        float s = lg;
        s += __shfl_xor(s, 1,  32);
        s += __shfl_xor(s, 2,  32);
        s += __shfl_xor(s, 4,  32);
        s += __shfl_xor(s, 8,  32);
        s += __shfl_xor(s, 16, 32);
        if (lane == 0) ysum[wave] = s;                 // wave 0 or 1
    }
    __syncthreads();
    if (t == 0) out[bIdx * Ldim + l] = ysum[0] + ysum[1];   // y[b,l], no atomics
}

// ---------------- fallback (workspace too small): convert in-kernel ----------------
__global__ __launch_bounds__(256) void mlp_fallback_kernel(
    const float* __restrict__ c_emb, const float* __restrict__ c_pred,
    const float* __restrict__ W1, const float* __restrict__ b1,
    const float* __restrict__ W2, const float* __restrict__ b2,
    float* __restrict__ out)
{
    __shared__ __align__(16) unsigned short Alf[32 * 40];
    __shared__ __align__(16) unsigned short Blf[Edim * 40];
    __shared__ float wl[32];

    const int t    = threadIdx.x;
    const int lane = t & 31;
    const int wave = t >> 5;
    const int wm   = wave >> 2;
    const int wn   = wave & 3;
    const int half = lane >> 4;
    const int l15  = lane & 15;
    const int l       = blockIdx.y;
    const int rowBase = blockIdx.x * 32;
    const int bIdx    = rowBase >> 6;

    if (t < 32) wl[t] = 0.0f;

    v8f acc[8];
    #pragma unroll
    for (int j = 0; j < 8; ++j) acc[j] = (v8f)0.0f;

    const float* __restrict__ W1l = W1 + (long)l * Edim * Edim;

    for (int k0 = 0; k0 < Edim; k0 += 32) {
        #pragma unroll
        for (int i = 0; i < 4; ++i) {
            int f = i * 256 + t;
            int m = f >> 5, k = f & 31;
            Alf[m * 40 + k] = f2bf(c_emb[(long)(rowBase + m) * Edim + k0 + k]);
        }
        #pragma unroll
        for (int i = 0; i < 16; ++i) {
            int f = i * 256 + t;
            int k = f >> 7, n = (f & 127) << 2;
            f32x4 w4 = *reinterpret_cast<const f32x4*>(&W1l[(long)(k0 + k) * Edim + n]);
            Blf[(n + 0) * 40 + k] = f2bf(w4.x);
            Blf[(n + 1) * 40 + k] = f2bf(w4.y);
            Blf[(n + 2) * 40 + k] = f2bf(w4.z);
            Blf[(n + 3) * 40 + k] = f2bf(w4.w);
        }
        __syncthreads();
        Frag af;
        const int m = wm * 16 + l15;
        af.q[0] = *reinterpret_cast<const i32x4*>(&Alf[m * 40 + half * 8]);
        af.q[1] = *reinterpret_cast<const i32x4*>(&Alf[m * 40 + 16 + half * 8]);
        #pragma unroll
        for (int j = 0; j < 8; ++j) {
            const int nt = (j * 4 + wn) * 16;
            Frag bfr;
            const i32x4* p = reinterpret_cast<const i32x4*>(&Blf[(nt + l15) * 40 + half * 8]);
            bfr.q[0] = p[0];
            bfr.q[1] = p[2];
            acc[j] = __builtin_amdgcn_wmma_f32_16x16x32_bf16(
                false, af.v, false, bfr.v, (short)0, acc[j], false, false);
        }
        __syncthreads();
    }

    float wsum[8];
    #pragma unroll
    for (int r = 0; r < 8; ++r) wsum[r] = 0.0f;
    #pragma unroll
    for (int j = 0; j < 8; ++j) {
        const int n  = (j * 4 + wn) * 16 + l15;
        const float bb = b1[l * Edim + n];
        const float ww = W2[l * Edim + n];
        #pragma unroll
        for (int r = 0; r < 8; ++r) {
            float v = acc[j][r] + bb;
            v = v > 0.0f ? v : 0.0f;
            wsum[r] += v * ww;
        }
    }
    #pragma unroll
    for (int r = 0; r < 8; ++r) {
        float s = wsum[r];
        s += __shfl_xor(s, 1, 32);
        s += __shfl_xor(s, 2, 32);
        s += __shfl_xor(s, 4, 32);
        s += __shfl_xor(s, 8, 32);
        if (l15 == 0) atomicAdd(&wl[wm * 16 + half * 8 + r], s);
    }
    __syncthreads();
    if (t < 32) {
        const int row   = rowBase + t;
        const float w   = wl[t] + b2[l];
        const float lg  = w * c_pred[row];
        out[Bdim * Ldim + (long)row * Ldim + l] = lg;
        float s = lg;
        s += __shfl_xor(s, 1,  32);
        s += __shfl_xor(s, 2,  32);
        s += __shfl_xor(s, 4,  32);
        s += __shfl_xor(s, 8,  32);
        s += __shfl_xor(s, 16, 32);
        if (t == 0) unsafeAtomicAdd(&out[bIdx * Ldim + l], s);
    }
}

extern "C" void kernel_launch(void* const* d_in, const int* in_sizes, int n_in,
                              void* d_out, int out_size, void* d_ws, size_t ws_size,
                              hipStream_t stream) {
    (void)in_sizes; (void)n_in; (void)out_size;
    const float* c_emb  = (const float*)d_in[0];
    const float* c_pred = (const float*)d_in[1];
    const float* W1     = (const float*)d_in[2];
    const float* b1     = (const float*)d_in[3];
    const float* W2     = (const float*)d_in[4];
    const float* b2     = (const float*)d_in[5];
    float* out = (float*)d_out;

    const size_t w1Bytes = (size_t)Ldim * Edim * Edim * sizeof(unsigned short); // 52.4 MB
    const size_t ceBytes = (size_t)Bdim * Cdim * Edim * sizeof(unsigned short); //  2.1 MB

    if (ws_size >= w1Bytes + ceBytes) {
        unsigned short* W1bf   = (unsigned short*)d_ws;
        unsigned short* cembbf = (unsigned short*)((char*)d_ws + w1Bytes);
        conv_w1_kernel<<<dim3(4, 16, Ldim), 256, 0, stream>>>(W1, W1bf);
        conv_cemb_kernel<<<(Bdim * Cdim * Edim / 4) / 256, 256, 0, stream>>>(c_emb, cembbf);
        mlp_pipelined_kernel<<<dim3(Bdim, Ldim), 256, 0, stream>>>(
            cembbf, c_pred, W1bf, b1, W2, b2, out);   // 32 b's x 100 labels
    } else {
        zero_y_kernel<<<(Bdim * Ldim + 255) / 256, 256, 0, stream>>>(out);
        mlp_fallback_kernel<<<dim3(64, Ldim), 256, 0, stream>>>(
            c_emb, c_pred, W1, b1, W2, b2, out);
    }
}
